// Mambablock_73710228734046
// MI455X (gfx1250) — compile-verified, matrix operand order checked by reading, not run on hardware
//
#include <hip/hip_runtime.h>
#include <hip/hip_bf16.h>
#include <math.h>

// ---------------- problem constants (from reference) ----------------
#define DIMX     1024
#define SEQX     2048
#define BATCHX   2
#define D_INNER  2048
#define D_STATE  16
#define DT_RANK  64
#define FFX      4096
#define ROWS     (BATCHX * SEQX)   // 4096
#define EPSX     1e-5f

// ---------------- CDNA5 WMMA types ----------------
typedef __attribute__((ext_vector_type(16))) __bf16 v16bf;
typedef __attribute__((ext_vector_type(8)))  float  v8f;

union FragBF { uint4 u[2]; v16bf v; };

__device__ __forceinline__ unsigned short f2bf(float f) {
  unsigned int u = __float_as_uint(f);
  u += 0x7FFFu + ((u >> 16) & 1u);     // round-to-nearest-even
  return (unsigned short)(u >> 16);
}
__device__ __forceinline__ float bf2f(unsigned short u) {
  return __uint_as_float((unsigned)u << 16);
}
__device__ __forceinline__ float silu_f(float x) { return x / (1.f + __expf(-x)); }

// ---------------- bf16 WMMA GEMM: C[M,N] = A[M,K] * W[N,K]^T ----------------
// Inputs are pre-converted bf16. Tiles staged via CDNA5 async copy
// (GLOBAL_LOAD_ASYNC_TO_LDS_B128, ASYNCcnt) with LDS double buffering.
// Requires: M % 128 == 0, K % 32 == 0 (true for all call sites).
// act: 1 = softplus ; out16: store bf16 instead of f32.
#define BM 128
#define BN 128
#define BK 32
#define LDSU 40                 // padded ushort row stride (80B): conflict-free b128 frags
#define AS_U (BM * LDSU)        // 5120 ushorts per tile buffer
#define SMEM_BYTES (4 * AS_U * 2)  // A0,B0,A1,B1 = 40960 bytes

__global__ void __launch_bounds__(256)
gemm_bf16_wmma(const unsigned short* __restrict__ A, int lda,
               const unsigned short* __restrict__ B, int ldb,
               float* __restrict__ C, int ldc,
               int M, int N, int K,
               const float* __restrict__ bias, int act, int out16)
{
  extern __shared__ unsigned short smem[];   // only LDS in this kernel -> offset 0

  const int tid  = threadIdx.x;
  const int lane = tid & 31;
  const int wave = tid >> 5;
  const int wm   = wave & 3;      // 4 wave-rows  x 32
  const int wn   = wave >> 2;     // 2 wave-cols  x 64
  const int m0   = blockIdx.y * BM;
  const int n0   = blockIdx.x * BN;

  const int lr  = tid >> 1;        // staging row 0..127
  const int lkc = (tid & 1) * 16;  // staging k-chunk: 16 bf16 = 32B
  const int nk  = K / BK;

  v8f acc[2][4] = {};

  // issue 4 async b128 copies per lane-slot (A 32B + W 32B) into buffer `buf`
  auto stage = [&](int buf, int k0) {
    const unsigned aOff = (unsigned)((2 * buf)     * AS_U + lr * LDSU + lkc) * 2u;
    const unsigned bOff = (unsigned)((2 * buf + 1) * AS_U + lr * LDSU + lkc) * 2u;
    const unsigned short* ga = A + (size_t)(m0 + lr) * lda + k0 + lkc;
    const int brow = (n0 + lr < N) ? (n0 + lr) : (N - 1);   // clamp -> uniform issue count
    const unsigned short* gb = B + (size_t)brow * ldb + k0 + lkc;
    asm volatile("global_load_async_to_lds_b128 %0, %1, off" :: "v"(aOff),       "v"(ga)     : "memory");
    asm volatile("global_load_async_to_lds_b128 %0, %1, off" :: "v"(aOff + 16u), "v"(ga + 8) : "memory");
    asm volatile("global_load_async_to_lds_b128 %0, %1, off" :: "v"(bOff),       "v"(gb)     : "memory");
    asm volatile("global_load_async_to_lds_b128 %0, %1, off" :: "v"(bOff + 16u), "v"(gb + 8) : "memory");
  };

  stage(0, 0);
  int cur = 0;
  for (int kt = 0; kt < nk; ++kt) {
    const bool more = (kt + 1) < nk;
    if (more) {
      stage(cur ^ 1, (kt + 1) * BK);
      asm volatile("s_wait_asynccnt 0x4" ::: "memory");  // current tile's 4 copies done
    } else {
      asm volatile("s_wait_asynccnt 0x0" ::: "memory");
    }
    __syncthreads();

    const unsigned aU = (unsigned)(2 * cur) * AS_U;
    const unsigned bU = aU + AS_U;
    FragBF a[2], b[4];
    {
      // 16-bit operand layout: lane<16 -> K0-7 & 16-23 ; lane>=16 -> K8-15 & 24-31
      const int l15 = lane & 15;
      const int kh  = (lane >> 4) * 8;
#pragma unroll
      for (int i = 0; i < 2; i++) {
        const int row = wm * 32 + i * 16 + l15;
        a[i].u[0] = *(const uint4*)&smem[aU + row * LDSU + kh];
        a[i].u[1] = *(const uint4*)&smem[aU + row * LDSU + 16 + kh];
      }
#pragma unroll
      for (int j = 0; j < 4; j++) {
        const int col = wn * 64 + j * 16 + l15;
        b[j].u[0] = *(const uint4*)&smem[bU + col * LDSU + kh];
        b[j].u[1] = *(const uint4*)&smem[bU + col * LDSU + 16 + kh];
      }
    }
#pragma unroll
    for (int i = 0; i < 2; i++)
#pragma unroll
      for (int j = 0; j < 4; j++)
        acc[i][j] = __builtin_amdgcn_wmma_f32_16x16x32_bf16(
            false, a[i].v, false, b[j].v, (short)0, acc[i][j], false, false);
    __syncthreads();
    cur ^= 1;
  }

  // epilogue: C/D layout lane<16: N=lane, M=vgpr ; lane>=16: N=lane-16, M=vgpr+8
  const int l15 = lane & 15;
  const int mh  = (lane >> 4) * 8;
#pragma unroll
  for (int i = 0; i < 2; i++) {
    const int mb = m0 + wm * 32 + i * 16 + mh;
#pragma unroll
    for (int j = 0; j < 4; j++) {
      const int n = n0 + wn * 64 + j * 16 + l15;
      if (n < N) {
        const float bv = bias ? bias[n] : 0.f;
#pragma unroll
        for (int v = 0; v < 8; v++) {
          const int m = mb + v;
          float val = acc[i][j][v] + bv;
          if (act == 1) val = (val > 20.f) ? val : log1pf(__expf(val));
          if (out16) ((unsigned short*)C)[(size_t)m * ldc + n] = f2bf(val);
          else       C[(size_t)m * ldc + n] = val;
        }
      }
    }
  }
}

// ---------------- f32 -> bf16 bulk convert (weights / activations) ----------
__global__ void __launch_bounds__(256)
cvt_bf16_kernel(const float* __restrict__ s, unsigned short* __restrict__ d, int n)
{
  const int i = blockIdx.x * 256 + threadIdx.x;
  if (i < n) d[i] = f2bf(s[i]);
}

// ---------------- residual accumulate + RMSNorm (row of 1024 per block) -----
__global__ void __launch_bounds__(256)
residual_rmsnorm_kernel(float* __restrict__ res, const float* __restrict__ hadd,
                        const float* __restrict__ w, unsigned short* __restrict__ out,
                        int add)
{
  __shared__ float red[256];
  const int tid = threadIdx.x;
  const size_t base = (size_t)blockIdx.x * DIMX;
  float v[4]; float ss = 0.f;
#pragma unroll
  for (int i = 0; i < 4; i++) {
    const int d = tid + i * 256;
    float r = hadd[base + d];
    if (add) r += res[base + d];
    v[i] = r; ss += r * r;
  }
#pragma unroll
  for (int i = 0; i < 4; i++) res[base + tid + i * 256] = v[i];
  red[tid] = ss; __syncthreads();
  for (int s = 128; s > 0; s >>= 1) { if (tid < s) red[tid] += red[tid + s]; __syncthreads(); }
  const float scale = rsqrtf(red[0] / (float)DIMX + EPSX);
#pragma unroll
  for (int i = 0; i < 4; i++) {
    const int d = tid + i * 256;
    out[base + d] = f2bf(v[i] * scale * w[d]);
  }
}

// ---------------- LayerNorm -> f32 (residual chain) + bf16 (GEMM input) -----
__global__ void __launch_bounds__(256)
layernorm_kernel(const float* __restrict__ x, const float* __restrict__ w,
                 const float* __restrict__ b, float* __restrict__ out,
                 unsigned short* __restrict__ out16)
{
  __shared__ float rs[256], rq[256];
  const int tid = threadIdx.x;
  const size_t base = (size_t)blockIdx.x * DIMX;
  float v[4]; float s = 0.f, q = 0.f;
#pragma unroll
  for (int i = 0; i < 4; i++) {
    const float r = x[base + tid + i * 256];
    v[i] = r; s += r; q += r * r;
  }
  rs[tid] = s; rq[tid] = q; __syncthreads();
  for (int t = 128; t > 0; t >>= 1) {
    if (tid < t) { rs[tid] += rs[tid + t]; rq[tid] += rq[tid + t]; }
    __syncthreads();
  }
  const float mu  = rs[0] / (float)DIMX;
  const float var = rq[0] / (float)DIMX - mu * mu;
  const float sc  = rsqrtf(var + EPSX);
#pragma unroll
  for (int i = 0; i < 4; i++) {
    const int d = tid + i * 256;
    const float r = (v[i] - mu) * sc * w[d] + b[d];
    out[base + d]   = r;
    out16[base + d] = f2bf(r);
  }
}

// ---------------- depthwise causal conv (K=4) + SiLU -> bf16 ----------------
__global__ void __launch_bounds__(256)
conv_silu_kernel(const float* __restrict__ xz, const float* __restrict__ cw,
                 const float* __restrict__ cb, unsigned short* __restrict__ xc)
{
  const int idx = blockIdx.x * 256 + threadIdx.x;
  const int e  = idx & (D_INNER - 1);
  const int l  = (idx >> 11) & (SEQX - 1);
  const int bb = idx >> 22;
  float acc = cb[e];
#pragma unroll
  for (int j = 0; j < 4; j++) {
    const int ls = l - 3 + j;
    if (ls >= 0) acc += cw[e * 4 + j] * xz[(size_t)(bb * SEQX + ls) * (2 * D_INNER) + e];
  }
  xc[(size_t)(bb * SEQX + l) * D_INNER + e] = f2bf(silu_f(acc));
}

// ---------------- selective scan (serial over L; thread = one channel) ------
// fuses y = scan + u*D, then y *= silu(z); emits bf16 for out_proj GEMM
__global__ void __launch_bounds__(256)
scan_kernel(const float* __restrict__ delta, const unsigned short* __restrict__ xc,
            const unsigned short* __restrict__ xdbl, const float* __restrict__ xz,
            const float* __restrict__ A_log, const float* __restrict__ Dp,
            unsigned short* __restrict__ yout)
{
  __shared__ float sBC[8 * 32];     // 8 timesteps x (16 B | 16 C)
  const int tid = threadIdx.x;
  const int b   = blockIdx.y;
  const int e   = blockIdx.x * 256 + tid;
  float Av[D_STATE], h[D_STATE];
#pragma unroll
  for (int n = 0; n < D_STATE; n++) {
    Av[n] = -__expf(A_log[(size_t)e * D_STATE + n]);
    h[n] = 0.f;
  }
  const float Dv = Dp[e];
  for (int lb = 0; lb < SEQX; lb += 8) {
    __syncthreads();
    {
      const int lc = tid >> 5, c = tid & 31;
      sBC[tid] = bf2f(xdbl[(size_t)(b * SEQX + lb + lc) * 96 + 64 + c]);
    }
    __syncthreads();
#pragma unroll
    for (int q = 0; q < 8; q++) {
      const int l = lb + q;
      const size_t idx = (size_t)(b * SEQX + l) * D_INNER + e;
      const float dt = delta[idx];
      const float u  = bf2f(xc[idx]);
      const float du = dt * u;
      float y = 0.f;
#pragma unroll
      for (int n = 0; n < D_STATE; n++) {
        h[n] = __expf(dt * Av[n]) * h[n] + du * sBC[q * 32 + n];
        y += h[n] * sBC[q * 32 + 16 + n];
      }
      y += u * Dv;
      const float z = xz[(size_t)(b * SEQX + l) * (2 * D_INNER) + D_INNER + e];
      yout[idx] = f2bf(y * silu_f(z));
    }
  }
}

// ---------------- GEGLU gate: v = a * gelu_exact(g) -> bf16 ----------------
__global__ void __launch_bounds__(256)
geglu_kernel(const float* __restrict__ u, unsigned short* __restrict__ v)
{
  const int idx = blockIdx.x * 256 + threadIdx.x;
  const int row = idx >> 12;           // / FFX
  const int f   = idx & (FFX - 1);
  const float a = u[(size_t)row * (2 * FFX) + f];
  const float g = u[(size_t)row * (2 * FFX) + FFX + f];
  v[(size_t)row * FFX + f] = f2bf(a * 0.5f * g * (1.f + erff(g * 0.70710678118654752f)));
}

// ---------------- host side ----------------
static inline void gemm(const unsigned short* A, int lda, const unsigned short* B, int ldb,
                        float* C, int ldc, int M, int N, int K,
                        const float* bias, int act, int out16, hipStream_t s)
{
  dim3 grid((unsigned)((N + BN - 1) / BN), (unsigned)((M + BM - 1) / BM));
  gemm_bf16_wmma<<<grid, 256, SMEM_BYTES, s>>>(A, lda, B, ldb, C, ldc, M, N, K, bias, act, out16);
}
static inline void cvt(const float* s, unsigned short* d, size_t n, hipStream_t st)
{
  cvt_bf16_kernel<<<(unsigned)((n + 255) / 256), 256, 0, st>>>(s, d, (int)n);
}

extern "C" void kernel_launch(void* const* d_in, const int* in_sizes, int n_in,
                              void* d_out, int out_size, void* d_ws, size_t ws_size,
                              hipStream_t stream)
{
  (void)in_sizes; (void)n_in; (void)out_size; (void)ws_size;
  const float* x      = (const float*)d_in[0];
  const float* in_w   = (const float*)d_in[1];
  const float* conv_w = (const float*)d_in[2];
  const float* conv_b = (const float*)d_in[3];
  const float* xp_w   = (const float*)d_in[4];
  const float* dtp_w  = (const float*)d_in[5];
  const float* dtp_b  = (const float*)d_in[6];
  const float* A_log  = (const float*)d_in[7];
  const float* Dp     = (const float*)d_in[8];
  const float* out_w  = (const float*)d_in[9];
  const float* rms_w  = (const float*)d_in[10];
  const float* ln_w   = (const float*)d_in[11];
  const float* ln_b   = (const float*)d_in[12];
  const float* ff_w1  = (const float*)d_in[13];
  const float* ff_b1  = (const float*)d_in[14];
  const float* ff_w2  = (const float*)d_in[15];
  const float* ff_b2  = (const float*)d_in[16];
  float* out = (float*)d_out;

  // ---- workspace layout ----
  float* ws = (float*)d_ws;
  size_t o = 0;
  float* resid = ws + o; o += (size_t)ROWS * DIMX;          //  4.19M f32
  float* xzbuf = ws + o; o += (size_t)ROWS * 2 * D_INNER;   // 16.78M
  float* delta = ws + o; o += (size_t)ROWS * D_INNER;       //  8.39M
  float* mout  = ws + o; o += (size_t)ROWS * DIMX;          //  4.19M  (sum = 33.55M)
  float* ffu   = (float*)d_ws;  // aliases [resid..mout) exactly: ROWS*2*FFX = 33.55M, dead by FFN
  float* hbuf  = ws + o; o += (size_t)ROWS * DIMX;

  unsigned short* us = (unsigned short*)(ws + o);
  size_t uo = 0;
  unsigned short* normed_bf = us + uo; uo += (size_t)ROWS * DIMX;
  unsigned short* xc_bf     = us + uo; uo += (size_t)ROWS * D_INNER;
  unsigned short* xdbl_bf   = us + uo; uo += (size_t)ROWS * 96;
  unsigned short* y_bf      = us + uo; uo += (size_t)ROWS * D_INNER;
  unsigned short* h_bf      = us + uo; uo += (size_t)ROWS * DIMX;
  unsigned short* v_bf      = us + uo; uo += (size_t)ROWS * FFX;
  unsigned short* w_in_bf   = us + uo; uo += (size_t)2 * 2 * D_INNER * DIMX;
  unsigned short* w_xp_bf   = us + uo; uo += (size_t)2 * 96 * D_INNER;
  unsigned short* w_dt_bf   = us + uo; uo += (size_t)2 * D_INNER * DT_RANK;
  unsigned short* w_out_bf  = us + uo; uo += (size_t)2 * DIMX * D_INNER;
  unsigned short* w_ff1_bf  = us + uo; uo += (size_t)2 * FFX * DIMX;
  unsigned short* w_ff2_bf  = us + uo; uo += (size_t)DIMX * FFX;

  // ---- one-time (per launch) weight conversion to bf16 ----
  cvt(in_w,  w_in_bf,  (size_t)2 * 2 * D_INNER * DIMX, stream);
  cvt(xp_w,  w_xp_bf,  (size_t)2 * 96 * D_INNER,       stream);
  cvt(dtp_w, w_dt_bf,  (size_t)2 * D_INNER * DT_RANK,  stream);
  cvt(out_w, w_out_bf, (size_t)2 * DIMX * D_INNER,     stream);
  cvt(ff_w1, w_ff1_bf, (size_t)2 * FFX * DIMX,         stream);
  cvt(ff_w2, w_ff2_bf, (size_t)DIMX * FFX,             stream);

  const float* hin = x;
  for (int i = 0; i < 2; i++) {
    residual_rmsnorm_kernel<<<ROWS, 256, 0, stream>>>(resid, hin, rms_w + (size_t)i * DIMX,
                                                      normed_bf, i);
    // in_proj: xz[4096,4096] = normed @ W^T
    gemm(normed_bf, DIMX, w_in_bf + (size_t)i * 2 * D_INNER * DIMX, DIMX,
         xzbuf, 2 * D_INNER, ROWS, 2 * D_INNER, DIMX, nullptr, 0, 0, stream);
    conv_silu_kernel<<<(BATCHX * SEQX * D_INNER) / 256, 256, 0, stream>>>(
        xzbuf, conv_w + (size_t)i * D_INNER * 4, conv_b + (size_t)i * D_INNER, xc_bf);
    // x_proj: xdbl[4096,96] (bf16 out) = xc @ W^T
    gemm(xc_bf, D_INNER, w_xp_bf + (size_t)i * 96 * D_INNER, D_INNER,
         (float*)xdbl_bf, 96, ROWS, 96, D_INNER, nullptr, 0, 1, stream);
    // dt_proj + bias + softplus: delta[4096,2048] = xdbl[:, :64] @ W^T
    gemm(xdbl_bf, 96, w_dt_bf + (size_t)i * D_INNER * DT_RANK, DT_RANK,
         delta, D_INNER, ROWS, D_INNER, DT_RANK, dtp_b + (size_t)i * D_INNER, 1, 0, stream);
    scan_kernel<<<dim3(D_INNER / 256, BATCHX), 256, 0, stream>>>(
        delta, xc_bf, xdbl_bf, xzbuf,
        A_log + (size_t)i * D_INNER * D_STATE, Dp + (size_t)i * D_INNER, y_bf);
    // out_proj: mout[4096,1024] = y @ W^T
    gemm(y_bf, D_INNER, w_out_bf + (size_t)i * DIMX * D_INNER, D_INNER,
         mout, DIMX, ROWS, DIMX, D_INNER, nullptr, 0, 0, stream);
    layernorm_kernel<<<ROWS, 256, 0, stream>>>(mout, ln_w + (size_t)i * DIMX,
                                               ln_b + (size_t)i * DIMX, hbuf, h_bf);
    hin = hbuf;
  }
  // FFN
  gemm(h_bf, DIMX, w_ff1_bf, DIMX, ffu, 2 * FFX, ROWS, 2 * FFX, DIMX, ff_b1, 0, 0, stream);
  geglu_kernel<<<((size_t)ROWS * FFX) / 256, 256, 0, stream>>>(ffu, v_bf);
  gemm(v_bf, FFX, w_ff2_bf, FFX, out, DIMX, ROWS, DIMX, FFX, ff_b2, 0, 0, stream);
}
